// AutoRegressiveGraphConvLayer_66331474919632
// MI455X (gfx1250) — matrix operand order repaired and checked by reading, not compiled, not run on hardware
//
#include <hip/hip_runtime.h>
#include <hip/hip_bf16.h>
#include <stdint.h>

// ---------------- static problem sizes (match reference) ----------------
constexpr int NN = 1024;   // nodes
constexpr int MM = 16;     // autoregressive window
constexpr int BB = 8;      // batch
constexpr int FN = 64, FE = 32;
constexpr int AN = 128, AE = 128;   // MLP out, hidden = 2*agg = 256
constexpr int ON = 64, OE = 32;
constexpr int NE = 16248;           // edges per batch
constexpr long RN = (long)BB * NE;  // total MLP rows = 129984 = 64*2031 exactly

typedef __bf16 bf16;
typedef __attribute__((ext_vector_type(4)))  bf16  bf16x4;
typedef __attribute__((ext_vector_type(8)))  bf16  bf16x8;
typedef __attribute__((ext_vector_type(16))) bf16  v16bf;
typedef __attribute__((ext_vector_type(8)))  float v8f;
typedef __attribute__((ext_vector_type(4)))  float f32x4;

union AFrag { v16bf v; bf16x8 h[2]; };

__device__ __forceinline__ v8f wmma_bf16(v16bf a, v16bf b, v8f c) {
  // D = A(16x32 bf16) x B(32x16 bf16) + C(16x16 f32)
  return __builtin_amdgcn_wmma_f32_16x16x32_bf16(false, a, false, b, (short)0, c,
                                                 false, false);
}

// Flat shared-pointer low 32 bits == wave-relative LDS byte address (ISA 10.2,
// LDS aperture uses the low 32 address bits).
__device__ __forceinline__ unsigned lds_off_of(const void* p) {
  return (unsigned)(uintptr_t)p;
}

// CDNA5 async global->LDS copy, 16B per lane; tracked by ASYNCcnt.
__device__ __forceinline__ void async_ld_b128(unsigned lds_off, const void* gaddr) {
  asm volatile("global_load_async_to_lds_b128 %0, %1, off"
               :: "v"(lds_off), "v"(gaddr) : "memory");
}
__device__ __forceinline__ void wait_async0() {
  asm volatile("s_wait_asynccnt 0" ::: "memory");
}

// ---------------- prep: f32 -> bf16 ----------------
__global__ void cvt_kernel(const float* __restrict__ src, bf16* __restrict__ dst,
                           int n) {
  int idx = blockIdx.x * blockDim.x + threadIdx.x;
  if (idx < n) dst[idx] = (bf16)src[idx];
}

// ---------------- node-path MLP: 96 -> 256 -> 128, scatter to node_agg ----
__global__ __launch_bounds__(128) void mlp_node_kernel(
    const bf16* __restrict__ nodesb, const bf16* __restrict__ edgesb,
    const bf16* __restrict__ W1, const bf16* __restrict__ W2,
    const float* __restrict__ b1, const float* __restrict__ b2,
    const int* __restrict__ node_src, const int* __restrict__ edge_dst,
    float* __restrict__ node_agg) {
  __shared__ bf16 Xs[64 * 96];     // 12 KB
  __shared__ bf16 Hs[64 * 256];    // 32 KB
  __shared__ bf16 W1s[96 * 256];   // 48 KB
  __shared__ int  dsts[64];
  const int tid = threadIdx.x;
  const long rowBase = (long)blockIdx.x * 64;

  // async stage W1 into LDS (no VGPR round-trip)
  for (int c = tid; c < 96 * 256 / 8; c += 128)
    async_ld_b128(lds_off_of(W1s + c * 8), W1 + c * 8);

  // async gather + concat: [nodes[src] | edges[e]] per row (RN is tile-exact)
  for (int c = tid; c < 64 * 12; c += 128) {
    int r = c / 12, ch = c % 12;
    long row = rowBase + r;
    long e = row % NE, b = row / NE;
    const bf16* g;
    if (ch < 8) g = nodesb + (b * NN + node_src[e]) * FN + ch * 8;
    else        g = edgesb + row * FE + (ch - 8) * 8;
    async_ld_b128(lds_off_of(Xs + r * 96 + ch * 8), g);
  }
  if (tid < 64) dsts[tid] = edge_dst[(int)((rowBase + tid) % NE)];
  wait_async0();
  __syncthreads();

  const int wave = tid >> 5, lane = tid & 31;
  const int col = lane & 15, half = lane >> 4;

  // phase 1: H = relu(X @ W1 + b1)
  const bf16* xrow = Xs + (wave * 16 + col) * 96;
  for (int nt = 0; nt < 16; ++nt) {
    const int n0 = nt * 16;
    v8f acc = {0.f, 0.f, 0.f, 0.f, 0.f, 0.f, 0.f, 0.f};
#pragma unroll
    for (int kk = 0; kk < 96; kk += 32) {
      AFrag a;
      a.h[0] = *(const bf16x8*)(xrow + kk + half * 8);
      a.h[1] = *(const bf16x8*)(xrow + kk + 16 + half * 8);
      v16bf bmat;
#pragma unroll
      for (int s = 0; s < 16; ++s)
        bmat[s] = W1s[(kk + (s < 8 ? half * 8 + s : 16 + half * 8 + s - 8)) * 256 + n0 + col];
      acc = wmma_bf16(a.v, bmat, acc);
    }
    float bias = b1[n0 + col];
#pragma unroll
    for (int r = 0; r < 8; ++r) {
      float v = acc[r] + bias;
      v = v > 0.f ? v : 0.f;
      Hs[(wave * 16 + half * 8 + r) * 256 + n0 + col] = (bf16)v;
    }
  }
  __syncthreads();

  // phase 2: A = relu(H @ W2 + b2), atomic scatter (W2 streamed, L2-hot)
  const bf16* hrow = Hs + (wave * 16 + col) * 256;
  for (int nt = 0; nt < 8; ++nt) {
    const int n0 = nt * 16;
    v8f acc = {0.f, 0.f, 0.f, 0.f, 0.f, 0.f, 0.f, 0.f};
#pragma unroll
    for (int kk = 0; kk < 256; kk += 32) {
      AFrag a;
      a.h[0] = *(const bf16x8*)(hrow + kk + half * 8);
      a.h[1] = *(const bf16x8*)(hrow + kk + 16 + half * 8);
      v16bf bmat;
#pragma unroll
      for (int s = 0; s < 16; ++s)
        bmat[s] = W2[(kk + (s < 8 ? half * 8 + s : 16 + half * 8 + s - 8)) * AN + n0 + col];
      acc = wmma_bf16(a.v, bmat, acc);
    }
    float bias = b2[n0 + col];
#pragma unroll
    for (int r = 0; r < 8; ++r) {
      int lr = wave * 16 + half * 8 + r;
      float v = acc[r] + bias;
      v = v > 0.f ? v : 0.f;
      long b = (rowBase + lr) / NE;
      atomicAdd(node_agg + (b * NN + dsts[lr]) * AN + n0 + col, v);
    }
  }
}

// ---------------- edge-path MLP: 160 -> 256 -> 128, write ae --------------
__global__ __launch_bounds__(128) void mlp_edge_kernel(
    const bf16* __restrict__ nodesb, const bf16* __restrict__ edgesb,
    const bf16* __restrict__ W1, const bf16* __restrict__ W2,
    const float* __restrict__ b1, const float* __restrict__ b2,
    const int* __restrict__ node_src, const int* __restrict__ edge_dst,
    float* __restrict__ ae) {
  __shared__ bf16 Xs[64 * 160];    // 20 KB
  __shared__ bf16 Hs[64 * 256];    // 32 KB
  __shared__ bf16 W1s[160 * 256];  // 80 KB
  const int tid = threadIdx.x;
  const long rowBase = (long)blockIdx.x * 64;

  for (int c = tid; c < 160 * 256 / 8; c += 128)
    async_ld_b128(lds_off_of(W1s + c * 8), W1 + c * 8);

  // async gather + concat: [nodes[src] | edges[e] | nodes[dst]]
  for (int c = tid; c < 64 * 20; c += 128) {
    int r = c / 20, ch = c % 20;
    long row = rowBase + r;
    long e = row % NE, b = row / NE;
    const bf16* g;
    if (ch < 8)       g = nodesb + (b * NN + node_src[e]) * FN + ch * 8;
    else if (ch < 12) g = edgesb + row * FE + (ch - 8) * 8;
    else              g = nodesb + (b * NN + edge_dst[e]) * FN + (ch - 12) * 8;
    async_ld_b128(lds_off_of(Xs + r * 160 + ch * 8), g);
  }
  wait_async0();
  __syncthreads();

  const int wave = tid >> 5, lane = tid & 31;
  const int col = lane & 15, half = lane >> 4;

  const bf16* xrow = Xs + (wave * 16 + col) * 160;
  for (int nt = 0; nt < 16; ++nt) {
    const int n0 = nt * 16;
    v8f acc = {0.f, 0.f, 0.f, 0.f, 0.f, 0.f, 0.f, 0.f};
#pragma unroll
    for (int kk = 0; kk < 160; kk += 32) {
      AFrag a;
      a.h[0] = *(const bf16x8*)(xrow + kk + half * 8);
      a.h[1] = *(const bf16x8*)(xrow + kk + 16 + half * 8);
      v16bf bmat;
#pragma unroll
      for (int s = 0; s < 16; ++s)
        bmat[s] = W1s[(kk + (s < 8 ? half * 8 + s : 16 + half * 8 + s - 8)) * 256 + n0 + col];
      acc = wmma_bf16(a.v, bmat, acc);
    }
    float bias = b1[n0 + col];
#pragma unroll
    for (int r = 0; r < 8; ++r) {
      float v = acc[r] + bias;
      v = v > 0.f ? v : 0.f;
      Hs[(wave * 16 + half * 8 + r) * 256 + n0 + col] = (bf16)v;
    }
  }
  __syncthreads();

  const bf16* hrow = Hs + (wave * 16 + col) * 256;
  for (int nt = 0; nt < 8; ++nt) {
    const int n0 = nt * 16;
    v8f acc = {0.f, 0.f, 0.f, 0.f, 0.f, 0.f, 0.f, 0.f};
#pragma unroll
    for (int kk = 0; kk < 256; kk += 32) {
      AFrag a;
      a.h[0] = *(const bf16x8*)(hrow + kk + half * 8);
      a.h[1] = *(const bf16x8*)(hrow + kk + 16 + half * 8);
      v16bf bmat;
#pragma unroll
      for (int s = 0; s < 16; ++s)
        bmat[s] = W2[(kk + (s < 8 ? half * 8 + s : 16 + half * 8 + s - 8)) * AE + n0 + col];
      acc = wmma_bf16(a.v, bmat, acc);
    }
    float bias = b2[n0 + col];
#pragma unroll
    for (int r = 0; r < 8; ++r) {
      long row = rowBase + wave * 16 + half * 8 + r;
      float v = acc[r] + bias;
      v = v > 0.f ? v : 0.f;
      ae[row * AE + n0 + col] = v;
    }
  }
}

// ---------------- final node linear: relu([agg*nrm | x] @ Wln + bln) ------
__global__ __launch_bounds__(256) void final_node_kernel(
    const float* __restrict__ node_agg, const bf16* __restrict__ nodesb,
    const bf16* __restrict__ Wl, const float* __restrict__ bl,
    const float* __restrict__ norm_node, float* __restrict__ out_nodes) {
  __shared__ bf16 Xs[128 * 192];   // 48 KB
  __shared__ bf16 Ws[192 * 64];    // 24 KB
  const int tid = threadIdx.x;
  const long rowBase = (long)blockIdx.x * 128;  // rows over BB*NN (exact)

  for (int c = tid; c < 192 * 64 / 8; c += 256)
    async_ld_b128(lds_off_of(Ws + c * 8), Wl + c * 8);

  // input-node half of the concat: async straight into LDS
  for (int c = tid; c < 128 * 8; c += 256) {
    int r = c / 8, ch = c % 8;
    long row = rowBase + r;
    async_ld_b128(lds_off_of(Xs + r * 192 + 128 + ch * 8),
                  nodesb + row * FN + ch * 8);
  }
  // normalized aggregate half: needs scaling + cvt, regular path
  for (int c = tid; c < 128 * 16; c += 256) {
    int r = c / 16, ch = c % 16;
    long row = rowBase + r;
    float nrm = norm_node[(int)(row % NN)];
    f32x4 a0 = *(const f32x4*)(node_agg + row * AN + ch * 8);
    f32x4 a1 = *(const f32x4*)(node_agg + row * AN + ch * 8 + 4);
    bf16x8 val;
#pragma unroll
    for (int j = 0; j < 4; ++j) {
      val[j] = (bf16)(a0[j] * nrm);
      val[4 + j] = (bf16)(a1[j] * nrm);
    }
    *(bf16x8*)(Xs + r * 192 + ch * 8) = val;
  }
  wait_async0();
  __syncthreads();

  const int wave = tid >> 5, lane = tid & 31;
  const int col = lane & 15, half = lane >> 4;
  const bf16* xrow = Xs + (wave * 16 + col) * 192;
  for (int nt = 0; nt < 4; ++nt) {
    const int n0 = nt * 16;
    v8f acc = {0.f, 0.f, 0.f, 0.f, 0.f, 0.f, 0.f, 0.f};
#pragma unroll
    for (int kk = 0; kk < 192; kk += 32) {
      AFrag a;
      a.h[0] = *(const bf16x8*)(xrow + kk + half * 8);
      a.h[1] = *(const bf16x8*)(xrow + kk + 16 + half * 8);
      v16bf bmat;
#pragma unroll
      for (int s = 0; s < 16; ++s)
        bmat[s] = Ws[(kk + (s < 8 ? half * 8 + s : 16 + half * 8 + s - 8)) * ON + n0 + col];
      acc = wmma_bf16(a.v, bmat, acc);
    }
    float bias = bl[n0 + col];
#pragma unroll
    for (int r = 0; r < 8; ++r) {
      float v = acc[r] + bias;
      v = v > 0.f ? v : 0.f;
      out_nodes[(rowBase + wave * 16 + half * 8 + r) * ON + n0 + col] = v;
    }
  }
}

// ------ final edge: per-node exclusive prefix of ae, norm, 160->32 linear --
__global__ __launch_bounds__(256) void final_edge_kernel(
    const float* __restrict__ ae, const bf16* __restrict__ edgesb,
    const bf16* __restrict__ Wl, const float* __restrict__ bl,
    const float* __restrict__ norm_edge, float* __restrict__ out_edges) {
  __shared__ bf16 Xs[8][16 * 160];  // 40 KB, one tile per wave
  __shared__ bf16 Ws[160 * 32];     // 10 KB
  const int tid = threadIdx.x;
  for (int c = tid; c < 160 * 32 / 8; c += 256)
    async_ld_b128(lds_off_of(Ws + c * 8), Wl + c * 8);
  wait_async0();
  __syncthreads();

  const int wave = tid >> 5, lane = tid & 31;
  const int p = blockIdx.x * 8 + wave;   // (b, i) block; uniform per wave
  if (p >= BB * (NN - 1)) return;
  const int b = p / (NN - 1);
  const int i = 1 + p % (NN - 1);
  const int t = i < MM ? i : MM;
  const int base = (i <= MM) ? (i * (i - 1)) / 2
                             : (MM * (MM - 1)) / 2 + (i - MM) * MM;
  bf16* xw = &Xs[wave][0];

  // exclusive prefix sum over the node's ae rows; lane owns 4 features
  f32x4 run = {0.f, 0.f, 0.f, 0.f};
  for (int pp = 0; pp < t; ++pp) {
    long e = base + pp;
    long row = (long)b * NE + e;
    float nrm = norm_edge[e];
    bf16x4 w;
#pragma unroll
    for (int j = 0; j < 4; ++j) w[j] = (bf16)(run[j] * nrm);
    *(bf16x4*)(xw + pp * 160 + lane * 4) = w;
    f32x4 v = *(const f32x4*)(ae + row * AE + lane * 4);
    run += v;
    xw[pp * 160 + 128 + lane] = edgesb[row * FE + lane];  // concat edge feats
  }
  for (int pp = t; pp < 16; ++pp) {
#pragma unroll
    for (int j = 0; j < 5; ++j) xw[pp * 160 + lane * 5 + j] = (bf16)0.f;
  }

  const int col = lane & 15, half = lane >> 4;
  const bf16* xrow = xw + col * 160;
  for (int nt = 0; nt < 2; ++nt) {
    const int n0 = nt * 16;
    v8f acc = {0.f, 0.f, 0.f, 0.f, 0.f, 0.f, 0.f, 0.f};
#pragma unroll
    for (int kk = 0; kk < 160; kk += 32) {
      AFrag a;
      a.h[0] = *(const bf16x8*)(xrow + kk + half * 8);
      a.h[1] = *(const bf16x8*)(xrow + kk + 16 + half * 8);
      v16bf bmat;
#pragma unroll
      for (int s = 0; s < 16; ++s)
        bmat[s] = Ws[(kk + (s < 8 ? half * 8 + s : 16 + half * 8 + s - 8)) * OE + n0 + col];
      acc = wmma_bf16(a.v, bmat, acc);
    }
    float bias = bl[n0 + col];
#pragma unroll
    for (int r = 0; r < 8; ++r) {
      int mr = half * 8 + r;
      if (mr < t) {
        float v = acc[r] + bias;
        v = v > 0.f ? v : 0.f;
        out_edges[((long)b * NE + base + mr) * OE + n0 + col] = v;
      }
    }
  }
}

// ---------------- launch ----------------
extern "C" void kernel_launch(void* const* d_in, const int* in_sizes, int n_in,
                              void* d_out, int out_size, void* d_ws, size_t ws_size,
                              hipStream_t stream) {
  const float* input_nodes = (const float*)d_in[0];
  const float* input_edges = (const float*)d_in[1];
  const float* W1n = (const float*)d_in[2];
  const float* b1n = (const float*)d_in[3];
  const float* W2n = (const float*)d_in[4];
  const float* b2n = (const float*)d_in[5];
  const float* Wln = (const float*)d_in[6];
  const float* bln = (const float*)d_in[7];
  const float* W1e = (const float*)d_in[8];
  const float* b1e = (const float*)d_in[9];
  const float* W2e = (const float*)d_in[10];
  const float* b2e = (const float*)d_in[11];
  const float* Wle = (const float*)d_in[12];
  const float* ble = (const float*)d_in[13];
  const int* node_src = (const int*)d_in[14];
  const int* edge_dst = (const int*)d_in[15];
  const float* norm_node = (const float*)d_in[18];
  const float* norm_edge = (const float*)d_in[19];

  float* out_nodes = (float*)d_out;
  float* out_edges = out_nodes + (size_t)BB * NN * ON;

  // workspace layout (256B aligned segments)
  char* ws = (char*)d_ws;
  size_t off = 0;
  auto alloc = [&](size_t bytes) -> void* {
    void* p = ws + off;
    off = (off + bytes + 255) & ~(size_t)255;
    return p;
  };
  const int n_nodesb = BB * NN * FN;
  const int n_edgesb = BB * NE * FE;
  bf16* nodesb = (bf16*)alloc((size_t)n_nodesb * 2);
  bf16* edgesb = (bf16*)alloc((size_t)n_edgesb * 2);
  bf16* W1nb = (bf16*)alloc((size_t)96 * 256 * 2);
  bf16* W2nb = (bf16*)alloc((size_t)256 * 128 * 2);
  bf16* Wlnb = (bf16*)alloc((size_t)192 * 64 * 2);
  bf16* W1eb = (bf16*)alloc((size_t)160 * 256 * 2);
  bf16* W2eb = (bf16*)alloc((size_t)256 * 128 * 2);
  bf16* Wleb = (bf16*)alloc((size_t)160 * 32 * 2);
  float* node_agg = (float*)alloc((size_t)BB * NN * AN * 4);
  float* ae = (float*)alloc((size_t)BB * NE * AE * 4);

  auto cvt = [&](const float* s, bf16* d, int n) {
    cvt_kernel<<<(n + 255) / 256, 256, 0, stream>>>(s, d, n);
  };
  cvt(input_nodes, nodesb, n_nodesb);
  cvt(input_edges, edgesb, n_edgesb);
  cvt(W1n, W1nb, 96 * 256);
  cvt(W2n, W2nb, 256 * 128);
  cvt(Wln, Wlnb, 192 * 64);
  cvt(W1e, W1eb, 160 * 256);
  cvt(W2e, W2eb, 256 * 128);
  cvt(Wle, Wleb, 160 * 32);

  hipMemsetAsync(node_agg, 0, (size_t)BB * NN * AN * 4, stream);

  const int gridMlp = (int)((RN + 63) / 64);  // 2031
  mlp_node_kernel<<<gridMlp, 128, 0, stream>>>(nodesb, edgesb, W1nb, W2nb, b1n,
                                               b2n, node_src, edge_dst, node_agg);
  mlp_edge_kernel<<<gridMlp, 128, 0, stream>>>(nodesb, edgesb, W1eb, W2eb, b1e,
                                               b2e, node_src, edge_dst, ae);
  final_node_kernel<<<(BB * NN) / 128, 256, 0, stream>>>(node_agg, nodesb, Wlnb,
                                                         bln, norm_node, out_nodes);
  final_edge_kernel<<<(BB * (NN - 1) + 7) / 8, 256, 0, stream>>>(ae, edgesb, Wleb,
                                                                 ble, norm_edge,
                                                                 out_edges);
  (void)in_sizes; (void)n_in; (void)out_size; (void)ws_size;
}
